// MultiHeadAttention_90812788507193
// MI455X (gfx1250) — matrix operand order repaired and checked by reading, not compiled, not run on hardware
//
#include <hip/hip_runtime.h>
#include <stdint.h>

// ---------------------------------------------------------------------------
// MI455X (gfx1250) multi-head attention forward.
// B=2, S=2048, D=1024, H=16, HD=64.  bf16 WMMA (v_wmma_f32_16x16x32_bf16),
// f32 accumulation, flash-attention streaming softmax.
//
// Pipeline:
//   0) one-time fp32->bf16 converts: x -> xb[M][D], W -> W^T bf16 (K-contiguous)
//   1) QKV GEMM  (4096x3072x1024), block tile 128x128, wave tile 32x64,
//      double-buffered LDS staging (global fill overlaps WMMA)
//   2) flash attention, 4 waves/block, 16 query rows/wave, 32-key tiles,
//      double-buffered K/V staging
//   3) proj GEMM (4096x1024x1024) -> fp32 out + bias
// ---------------------------------------------------------------------------

#define BB   2
#define SS   2048
#define DD   1024
#define HH   16
#define HDD  64
#define M_TOT (BB * SS)          // 4096
#define N_QKV (3 * DD)           // 3072

typedef __attribute__((ext_vector_type(16))) __bf16        v16bf;
typedef __attribute__((ext_vector_type(8)))  float         v8f;
typedef __attribute__((ext_vector_type(4)))  unsigned int  u32x4;
typedef __attribute__((ext_vector_type(2)))  float         f32x2;

// fp32 -> bf16 bits, round-to-nearest-even
__device__ __forceinline__ unsigned short f2bf(float f) {
    union { float f; unsigned int u; } c; c.f = f;
    unsigned int u = c.u;
    return (unsigned short)((u + 0x7FFFu + ((u >> 16) & 1u)) >> 16);
}

union FragBF { v16bf v; u32x4 q[2]; };
union Pack8  { u32x4 q; unsigned short s[8]; };

// A-fragment (16x32 bf16, MxK) from a row-major [rows][ld] bf16(ushort) tile.
// ISA: lanes 0-15: row=lane, K={0..7,16..23}; lanes 16-31: row=lane&15, K={8..15,24..31}
__device__ __forceinline__ v16bf load_a_frag(const unsigned short* tile, int ld, int kBase) {
    const int lane = threadIdx.x & 31;
    const int row  = lane & 15;
    const int k0   = kBase + ((lane & 16) ? 8 : 0);
    const unsigned short* p = tile + row * ld + k0;
    FragBF r;
    r.q[0] = *reinterpret_cast<const u32x4*>(p);
    r.q[1] = *reinterpret_cast<const u32x4*>(p + 16);
    return r.v;
}

// B-fragment (32x16 bf16, KxN) from a transposed [N][ld(K)] bf16(ushort) tile.
// Lanes 0-15: N=lane, K={0..7,16..23}; lanes 16-31: K={8..15,24..31}
// -> two 16-byte reads per lane at [nBase+lane%16][klo] and [..][klo+16].
__device__ __forceinline__ v16bf load_b_frag(const unsigned short* tileT, int ld, int nBase) {
    const int lane = threadIdx.x & 31;
    const int n    = nBase + (lane & 15);
    const int klo  = (lane & 16) ? 8 : 0;
    const unsigned short* p = tileT + n * ld + klo;
    FragBF r;
    r.q[0] = *reinterpret_cast<const u32x4*>(p);
    r.q[1] = *reinterpret_cast<const u32x4*>(p + 16);
    return r.v;
}

#define WMMA_BF16(a, b, c) \
    __builtin_amdgcn_wmma_f32_16x16x32_bf16(false, (a), false, (b), (short)0, (c), false, false)

// ---------------------------------------------------------------------------
// Kernel 0a: fp32 -> bf16 copy (2 elements / thread, packed u32 stores)
// ---------------------------------------------------------------------------
__global__ __launch_bounds__(256) void cvt_bf16_kernel(
    const float* __restrict__ src, unsigned int* __restrict__ dst, int n2) {
    int i = blockIdx.x * blockDim.x + threadIdx.x;
    if (i < n2) {
        f32x2 f = *reinterpret_cast<const f32x2*>(src + 2 * (size_t)i);
        dst[i] = (unsigned int)f2bf(f.x) | ((unsigned int)f2bf(f.y) << 16);
    }
}

// ---------------------------------------------------------------------------
// Kernel 0b: fp32 W[K][N] -> bf16 W^T[N][K]  (one-time transpose+convert)
// ---------------------------------------------------------------------------
__global__ __launch_bounds__(256) void cvt_wT_kernel(
    const float* __restrict__ W, unsigned short* __restrict__ Wt, int K, int N) {
    int i = blockIdx.x * blockDim.x + threadIdx.x;
    if (i < K * N) {
        int k = i / N, n = i % N;
        Wt[(size_t)n * K + k] = f2bf(W[i]);
    }
}

// ---------------------------------------------------------------------------
// GEMM tile staging: 128 rows x 32 k (A) and 128 n x 32 k (B^T), b128 copies.
// ---------------------------------------------------------------------------
__device__ __forceinline__ void stage_gemm_tiles(
    const unsigned short* __restrict__ Ag, const unsigned short* __restrict__ Btg,
    int mBase, int nBase, int Kdim, int kb,
    unsigned short (*At)[32], unsigned short (*Bt)[32]) {
    const int tid = threadIdx.x;
    #pragma unroll
    for (int i = 0; i < 2; ++i) {
        int idx = tid + i * 256;
        int r = idx >> 2, c = (idx & 3) * 8;
        *reinterpret_cast<u32x4*>(&At[r][c]) =
            *reinterpret_cast<const u32x4*>(Ag + (size_t)(mBase + r) * Kdim + kb + c);
        *reinterpret_cast<u32x4*>(&Bt[r][c]) =
            *reinterpret_cast<const u32x4*>(Btg + (size_t)(nBase + r) * Kdim + kb + c);
    }
}

// ---------------------------------------------------------------------------
// Double-buffered GEMM mainloop: C[128x128] block tile, 8 waves (4M x 2N),
// wave tile 32x64.  One barrier per K-step; stage(k+1) overlaps compute(k).
// ---------------------------------------------------------------------------
__device__ __forceinline__ void gemm_mainloop(
    const unsigned short* __restrict__ Ag, const unsigned short* __restrict__ Btg,
    int mBase, int nBase, int Kdim,
    unsigned short (*At)[128][32], unsigned short (*Bt)[128][32], v8f acc[2][4]) {
    const int wave  = threadIdx.x >> 5;
    const int waveM = wave >> 1;               // 0..3 -> 32-row band
    const int waveN = wave & 1;                // 0..1 -> 64-col band
    const int nIter = Kdim / 32;

    stage_gemm_tiles(Ag, Btg, mBase, nBase, Kdim, 0, At[0], Bt[0]);
    for (int it = 0; it < nIter; ++it) {
        __syncthreads();                       // buf[cur] staged; buf[nxt] free
        const int cur = it & 1;
        if (it + 1 < nIter) {
            stage_gemm_tiles(Ag, Btg, mBase, nBase, Kdim, (it + 1) * 32,
                             At[1 - cur], Bt[1 - cur]);
            if (it + 2 < nIter)                // speculative prefetch of tile k+2
                __builtin_prefetch(Ag + (size_t)(mBase + (threadIdx.x >> 1)) * Kdim
                                      + (it + 2) * 32, 0, 0);
        }
        v16bf a0 = load_a_frag(&At[cur][waveM * 32][0],      32, 0);
        v16bf a1 = load_a_frag(&At[cur][waveM * 32 + 16][0], 32, 0);
        #pragma unroll
        for (int ni = 0; ni < 4; ++ni) {
            v16bf b = load_b_frag(&Bt[cur][0][0], 32, waveN * 64 + ni * 16);
            acc[0][ni] = WMMA_BF16(a0, b, acc[0][ni]);
            acc[1][ni] = WMMA_BF16(a1, b, acc[1][ni]);
        }
    }
}

// ---------------------------------------------------------------------------
// Kernel 1: QKV GEMM, scatter bf16 q/k/v in [B, H, S, HD] layout.
// ---------------------------------------------------------------------------
__global__ __launch_bounds__(256) void qkv_gemm_kernel(
    const unsigned short* __restrict__ xb, const unsigned short* __restrict__ WqkvT,
    const float* __restrict__ bqkv,
    unsigned short* __restrict__ q, unsigned short* __restrict__ k,
    unsigned short* __restrict__ v) {
    __shared__ unsigned short At[2][128][32];
    __shared__ unsigned short Bt[2][128][32];
    const int lane  = threadIdx.x & 31;
    const int wave  = threadIdx.x >> 5;
    const int waveM = wave >> 1, waveN = wave & 1;
    const int mBase = blockIdx.x * 128;
    const int nBase = blockIdx.y * 128;

    v8f acc[2][4] = {};
    gemm_mainloop(xb, WqkvT, mBase, nBase, DD, At, Bt, acc);

    const int rowOff = (lane & 16) ? 8 : 0;
    #pragma unroll
    for (int ni = 0; ni < 4; ++ni) {
        const int nCol  = nBase + waveN * 64 + ni * 16 + (lane & 15);
        const float bias = bqkv[nCol];
        const int which = nCol / DD;
        const int dIdx  = nCol % DD;
        const int head  = dIdx / HDD;
        const int hd    = dIdx % HDD;
        unsigned short* dst = (which == 0) ? q : ((which == 1) ? k : v);
        #pragma unroll
        for (int mi = 0; mi < 2; ++mi) {
            #pragma unroll
            for (int j = 0; j < 8; ++j) {
                int m  = mBase + waveM * 32 + mi * 16 + j + rowOff;
                int bb = m / SS, ss = m % SS;
                dst[(((size_t)bb * HH + head) * SS + ss) * HDD + hd] =
                    f2bf(acc[mi][ni][j] + bias);
            }
        }
    }
}

// ---------------------------------------------------------------------------
// Attention K/V tile staging: K row-major [key][d], V transposed [hd][key].
// ---------------------------------------------------------------------------
__device__ __forceinline__ void stage_kv_tiles(
    const unsigned short* __restrict__ kg, const unsigned short* __restrict__ vg,
    size_t kvOff, unsigned short (*Kt)[64], unsigned short (*Vt)[32]) {
    const int tid = threadIdx.x;
    #pragma unroll
    for (int i = 0; i < 2; ++i) {
        int idx = tid + i * 128;               // 256 x 16B chunks
        int r = idx >> 3, c = (idx & 7) * 8;
        *reinterpret_cast<u32x4*>(&Kt[r][c]) =
            *reinterpret_cast<const u32x4*>(kg + kvOff + (size_t)r * HDD + c);
        Pack8 d;
        d.q = *reinterpret_cast<const u32x4*>(vg + kvOff + (size_t)r * HDD + c);
        #pragma unroll
        for (int e = 0; e < 8; ++e) Vt[c + e][r] = d.s[e];
    }
}

// ---------------------------------------------------------------------------
// Kernel 2: flash attention.  Grid: (S/64, B*H), 128 thr (4 waves).
// ---------------------------------------------------------------------------
__global__ __launch_bounds__(128) void attn_kernel(
    const unsigned short* __restrict__ qg, const unsigned short* __restrict__ kg,
    const unsigned short* __restrict__ vg, unsigned short* __restrict__ og) {
    __shared__ unsigned short Kt[2][32][64];     // [key][d]
    __shared__ unsigned short Vt[2][64][32];     // [hd][key] (transposed)
    __shared__ unsigned short Pst[4][16][32];    // per-wave P staging

    const int tid    = threadIdx.x;
    const int lane   = tid & 31;
    const int wave   = tid >> 5;
    const int qTile  = blockIdx.x;
    const int bh     = blockIdx.y;
    const size_t baseBH = (size_t)bh * SS * HDD;
    const int qBase  = qTile * 64 + wave * 16;
    const int rowOff = (lane & 16) ? 8 : 0;
    const int nIdx   = lane & 15;

    // Q A-fragments, two 32-wide d-chunks, straight from global
    v16bf qa[2];
    {
        const unsigned short* qrow = qg + baseBH + (size_t)(qBase + (lane & 15)) * HDD;
        const int k0 = (lane & 16) ? 8 : 0;
        #pragma unroll
        for (int dc = 0; dc < 2; ++dc) {
            FragBF r;
            r.q[0] = *reinterpret_cast<const u32x4*>(qrow + dc * 32 + k0);
            r.q[1] = *reinterpret_cast<const u32x4*>(qrow + dc * 32 + k0 + 16);
            qa[dc] = r.v;
        }
    }

    v8f o0 = {}, o1 = {}, o2 = {}, o3 = {};
    float mrow[8], lrow[8];
    #pragma unroll
    for (int j = 0; j < 8; ++j) { mrow[j] = -1e30f; lrow[j] = 0.0f; }

    const int numKT = qTile * 2 + 2;
    stage_kv_tiles(kg, vg, baseBH, Kt[0], Vt[0]);
    for (int kt = 0; kt < numKT; ++kt) {
        __syncthreads();                         // buf[cur] staged
        const int cur = kt & 1;
        if (kt + 1 < numKT)
            stage_kv_tiles(kg, vg, baseBH + (size_t)(kt + 1) * 32 * HDD,
                           Kt[1 - cur], Vt[1 - cur]);

        // S = Q K^T over this 32-key tile
        v8f s0 = {}, s1 = {};
        #pragma unroll
        for (int dc = 0; dc < 2; ++dc) {
            v16bf b0 = load_b_frag(&Kt[cur][0][dc * 32], 64, 0);
            v16bf b1 = load_b_frag(&Kt[cur][0][dc * 32], 64, 16);
            s0 = WMMA_BF16(qa[dc], b0, s0);
            s1 = WMMA_BF16(qa[dc], b1, s1);
        }

        // scale + causal mask + online softmax
        const float scale = 0.125f;              // HD^-0.5
        float alpha[8];
        #pragma unroll
        for (int j = 0; j < 8; ++j) {
            int qi  = qBase + j + rowOff;
            int kj0 = kt * 32 + nIdx;
            float v0 = s0[j] * scale; if (kj0      > qi) v0 = -1e30f;
            float v1 = s1[j] * scale; if (kj0 + 16 > qi) v1 = -1e30f;
            float mx = fmaxf(v0, v1);
            #pragma unroll
            for (int off = 1; off < 16; off <<= 1) mx = fmaxf(mx, __shfl_xor(mx, off, 32));
            float mnew = fmaxf(mrow[j], mx);
            alpha[j] = __expf(mrow[j] - mnew);
            mrow[j]  = mnew;
            float p0 = __expf(v0 - mnew);
            float p1 = __expf(v1 - mnew);
            float sum = p0 + p1;
            #pragma unroll
            for (int off = 1; off < 16; off <<= 1) sum += __shfl_xor(sum, off, 32);
            lrow[j] = lrow[j] * alpha[j] + sum;
            s0[j] = p0; s1[j] = p1;
        }

        // P: C layout -> per-wave LDS patch -> A fragment (in-order DS in wave)
        #pragma unroll
        for (int j = 0; j < 8; ++j) {
            int r = j + rowOff;
            Pst[wave][r][nIdx]      = f2bf(s0[j]);
            Pst[wave][r][nIdx + 16] = f2bf(s1[j]);
        }
        #pragma unroll
        for (int j = 0; j < 8; ++j) {
            o0[j] *= alpha[j]; o1[j] *= alpha[j];
            o2[j] *= alpha[j]; o3[j] *= alpha[j];
        }
        v16bf pa = load_a_frag(&Pst[wave][0][0], 32, 0);
        o0 = WMMA_BF16(pa, load_b_frag(&Vt[cur][0][0], 32, 0),  o0);
        o1 = WMMA_BF16(pa, load_b_frag(&Vt[cur][0][0], 32, 16), o1);
        o2 = WMMA_BF16(pa, load_b_frag(&Vt[cur][0][0], 32, 32), o2);
        o3 = WMMA_BF16(pa, load_b_frag(&Vt[cur][0][0], 32, 48), o3);
    }

    // Normalize, store bf16 attention output token-major [B*S][D]
    const int b = bh / HH, h = bh % HH;
    #pragma unroll
    for (int j = 0; j < 8; ++j) {
        float inv = 1.0f / lrow[j];
        int s = qBase + j + rowOff;
        unsigned short* orow = og + ((size_t)b * SS + s) * DD + h * HDD;
        orow[nIdx]      = f2bf(o0[j] * inv);
        orow[nIdx + 16] = f2bf(o1[j] * inv);
        orow[nIdx + 32] = f2bf(o2[j] * inv);
        orow[nIdx + 48] = f2bf(o3[j] * inv);
    }
}

// ---------------------------------------------------------------------------
// Kernel 3: out = attn @ W_proj + b_proj -> fp32
// ---------------------------------------------------------------------------
__global__ __launch_bounds__(256) void proj_gemm_kernel(
    const unsigned short* __restrict__ attn, const unsigned short* __restrict__ WpT,
    const float* __restrict__ bp, float* __restrict__ out) {
    __shared__ unsigned short At[2][128][32];
    __shared__ unsigned short Bt[2][128][32];
    const int lane  = threadIdx.x & 31;
    const int wave  = threadIdx.x >> 5;
    const int waveM = wave >> 1, waveN = wave & 1;
    const int mBase = blockIdx.x * 128;
    const int nBase = blockIdx.y * 128;

    v8f acc[2][4] = {};
    gemm_mainloop(attn, WpT, mBase, nBase, DD, At, Bt, acc);

    const int rowOff = (lane & 16) ? 8 : 0;
    #pragma unroll
    for (int ni = 0; ni < 4; ++ni) {
        const int nCol  = nBase + waveN * 64 + ni * 16 + (lane & 15);
        const float bias = bp[nCol];
        #pragma unroll
        for (int mi = 0; mi < 2; ++mi) {
            #pragma unroll
            for (int j = 0; j < 8; ++j) {
                int m = mBase + waveM * 32 + mi * 16 + j + rowOff;
                out[(size_t)m * DD + nCol] = acc[mi][ni][j] + bias;
            }
        }
    }
}

// ---------------------------------------------------------------------------
extern "C" void kernel_launch(void* const* d_in, const int* in_sizes, int n_in,
                              void* d_out, int out_size, void* d_ws, size_t ws_size,
                              hipStream_t stream) {
    const float* x     = (const float*)d_in[0];
    const float* Wqkv  = (const float*)d_in[1];
    const float* bqkv  = (const float*)d_in[2];
    const float* Wproj = (const float*)d_in[3];
    const float* bproj = (const float*)d_in[4];
    float* out = (float*)d_out;

    const size_t TEN = (size_t)BB * SS * DD;     // 4,194,304 elems
    unsigned short* xb    = (unsigned short*)d_ws;          // 8 MB
    unsigned short* WqkvT = xb + TEN;                       // 6 MB  [3D][D]
    unsigned short* WprojT= WqkvT + (size_t)DD * N_QKV;     // 2 MB  [D][D]
    unsigned short* q     = WprojT + (size_t)DD * DD;       // 8 MB each
    unsigned short* k     = q + TEN;
    unsigned short* v     = k + TEN;
    unsigned short* attn  = v + TEN;                        // total 48 MB

    // one-time conversions
    cvt_bf16_kernel<<<(int)(TEN / 2 / 256), 256, 0, stream>>>(
        x, (unsigned int*)xb, (int)(TEN / 2));
    cvt_wT_kernel<<<(DD * N_QKV + 255) / 256, 256, 0, stream>>>(
        Wqkv, WqkvT, DD, N_QKV);
    cvt_wT_kernel<<<(DD * DD + 255) / 256, 256, 0, stream>>>(
        Wproj, WprojT, DD, DD);

    qkv_gemm_kernel<<<dim3(M_TOT / 128, N_QKV / 128), 256, 0, stream>>>(
        xb, WqkvT, bqkv, q, k, v);
    attn_kernel<<<dim3(SS / 64, BB * HH), 128, 0, stream>>>(q, k, v, attn);
    proj_gemm_kernel<<<dim3(M_TOT / 128, DD / 128), 256, 0, stream>>>(
        attn, WprojT, bproj, out);
}
